// SpikeTDANetLayer_72610717106761
// MI455X (gfx1250) — compile-verified, matrix operand order checked by reading, not run on hardware
//
#include <hip/hip_runtime.h>

typedef __attribute__((ext_vector_type(16))) __bf16 v16bf;
typedef __attribute__((ext_vector_type(8)))  float  v8f;

#define TT   8
#define CC   128
#define HH   4
#define DHD  32
#define WWIN 4
#define KKD  5
#define TAUF 0.95f
#define VTHF 1.0f

// ---------------- helpers ----------------

__device__ __forceinline__ unsigned short f2bf_bits(float f) {
  unsigned u = __builtin_bit_cast(unsigned, f);
  unsigned r = u + 0x7FFFu + ((u >> 16) & 1u);   // round-to-nearest-even bf16
  return (unsigned short)(r >> 16);
}

__device__ __forceinline__ float waveRed(float v) {
#pragma unroll
  for (int m = 16; m > 0; m >>= 1) v += __shfl_xor(v, m, 32);
  return v;
}

__device__ __forceinline__ float gelu_tanh(float x) {
  float x3 = x * x * x;
  return 0.5f * x * (1.0f + tanhf(0.7978845608028654f * (x + 0.044715f * x3)));
}

// CDNA5 16-bit A/B fragment (16x32 tile, row-major source with leading dim ld):
// lane<16: row=lane, K = {k0..k0+7, k0+16..k0+23}
// lane>=16: row=lane-16, K = {k0+8..k0+15, k0+24..k0+31}
__device__ __forceinline__ v16bf load_frag(const unsigned short* base, int ld, int k0, int lane) {
  int r = lane & 15;
  const unsigned short* p = base + r * ld + k0 + ((lane & 16) ? 8 : 0);
  v16bf f;
#pragma unroll
  for (int i = 0; i < 8; ++i) f[i]     = __builtin_bit_cast(__bf16, p[i]);
#pragma unroll
  for (int i = 0; i < 8; ++i) f[8 + i] = __builtin_bit_cast(__bf16, p[16 + i]);
  return f;
}

// ---------------- utility kernels ----------------

__global__ void kzero(float* __restrict__ p, long long n4) {
  long long i = (long long)blockIdx.x * blockDim.x + threadIdx.x;
  if (i < n4) ((float4*)p)[i] = make_float4(0.f, 0.f, 0.f, 0.f);
}

__global__ void kw2bf(const float* __restrict__ W, unsigned short* __restrict__ Wt,
                      int Kd, int Nd) {  // W[Kd,Nd] -> Wt[Nd,Kd] bf16
  int i = blockIdx.x * blockDim.x + threadIdx.x;
  if (i >= Kd * Nd) return;
  int k = i / Nd, n = i - k * Nd;
  Wt[n * Kd + k] = f2bf_bits(W[i]);
}

__global__ void kdeg(const int* __restrict__ dst, float* __restrict__ deg, int E) {
  int e = blockIdx.x * blockDim.x + threadIdx.x;
  if (e < E) atomicAdd(&deg[dst[e]], 1.0f);
}

__global__ void kinv(float* __restrict__ deg, int n) {
  int i = blockIdx.x * blockDim.x + threadIdx.x;
  if (i < n) deg[i] = 1.0f / fmaxf(deg[i], 1.0f);
}

// wave per (t,e): gather one source row (lane x float4), atomic-scatter to dst row
__global__ void kscatter(const float* __restrict__ xin, const int* __restrict__ src,
                         const int* __restrict__ dst, const float* __restrict__ gate,
                         float* __restrict__ agg, int E, int Nn) {
  int gw = blockIdx.x * (blockDim.x >> 5) + (threadIdx.x >> 5);
  if (gw >= TT * E) return;
  int t = gw / E, e = gw - t * E;
  int s = src[e], d = dst[e];
  int lane = threadIdx.x & 31;
  float4 v = ((const float4*)(xin + ((size_t)t * Nn + s) * CC))[lane];
  if (gate) {
    float g = gate[t * Nn + s];
    v.x *= g; v.y *= g; v.z *= g; v.w *= g;
  }
  float* o = agg + ((size_t)t * Nn + d) * CC + lane * 4;
  atomicAdd(o + 0, v.x); atomicAdd(o + 1, v.y);
  atomicAdd(o + 2, v.z); atomicAdd(o + 3, v.w);
}

// ---------------- WMMA GEMM: out[M,128] = (rowScale*A)[M,128] @ Bt^T + bias ----
// Bt is bf16 [128 outcols, 128 k] (pre-transposed). In-place (A==out) safe.
__global__ void kgemm128(const float* __restrict__ A, const unsigned short* __restrict__ Bt,
                         const float* __restrict__ bias, const float* __restrict__ rowScale,
                         int scaleMod, float* __restrict__ out, int M) {
  __shared__ unsigned short aT[16 * 136];
  int rowBase = blockIdx.x * 16;
  for (int i = threadIdx.x; i < 16 * CC; i += blockDim.x) {
    int r = i >> 7, c = i & 127;
    float v = A[(size_t)(rowBase + r) * CC + c];
    if (rowScale) v *= rowScale[(rowBase + r) % scaleMod];
    aT[r * 136 + c] = f2bf_bits(v);
  }
  __syncthreads();
  int wave = threadIdx.x >> 5, lane = threadIdx.x & 31;
  int c0 = wave * 16;
  v8f acc = {};
#pragma unroll
  for (int k0 = 0; k0 < CC; k0 += 32) {
    v16bf a = load_frag(aT, 136, k0, lane);
    v16bf b = load_frag(Bt + (size_t)c0 * CC, CC, k0, lane);
    acc = __builtin_amdgcn_wmma_f32_16x16x32_bf16(false, a, false, b, (short)0, acc,
                                                  false, false);
  }
  int col = c0 + (lane & 15);
  int rOff = (lane & 16) ? 8 : 0;
  float bv = bias[col];
#pragma unroll
  for (int r = 0; r < 8; ++r)
    out[(size_t)(rowBase + rOff + r) * CC + col] = acc[r] + bv;
}

// ---------------- LayerNorm(a [+ b]) -> out, wave per 128-wide row -----------
__global__ void klnres(const float* __restrict__ a, const float* __restrict__ b,
                       const float* __restrict__ g, const float* __restrict__ be,
                       float* __restrict__ out, int Mrows) {
  int wid = blockIdx.x * (blockDim.x >> 5) + (threadIdx.x >> 5);
  if (wid >= Mrows) return;
  int lane = threadIdx.x & 31;
  float4 v = ((const float4*)(a + (size_t)wid * CC))[lane];
  if (b) {
    float4 u = ((const float4*)(b + (size_t)wid * CC))[lane];
    v.x += u.x; v.y += u.y; v.z += u.z; v.w += u.w;
  }
  float s  = waveRed(v.x + v.y + v.z + v.w);
  float sq = waveRed(v.x * v.x + v.y * v.y + v.z * v.z + v.w * v.w);
  float mean = s * (1.0f / CC);
  float var  = sq * (1.0f / CC) - mean * mean;
  float rstd = rsqrtf(var + 1e-5f);
  float4 gg = ((const float4*)g)[lane];
  float4 bb = ((const float4*)be)[lane];
  float4 o;
  o.x = (v.x - mean) * rstd * gg.x + bb.x;
  o.y = (v.y - mean) * rstd * gg.y + bb.y;
  o.z = (v.z - mean) * rstd * gg.z + bb.z;
  o.w = (v.w - mean) * rstd * gg.w + bb.w;
  ((float4*)(out + (size_t)wid * CC))[lane] = o;
}

// ------------- causal depthwise delay conv + residual + LN, wave per row -----
__global__ void kdelayln(const float* __restrict__ x1, const float* __restrict__ dk,
                         const float* __restrict__ g, const float* __restrict__ be,
                         float* __restrict__ out, int Nn) {
  int wid = blockIdx.x * (blockDim.x >> 5) + (threadIdx.x >> 5);
  if (wid >= TT * Nn) return;
  int lane = threadIdx.x & 31;
  int t = wid / Nn, n = wid - t * Nn;
  int c = lane * 4;
  float4 v = *(const float4*)(x1 + ((size_t)t * Nn + n) * CC + c);  // residual x1[t]
#pragma unroll
  for (int j = 0; j < KKD; ++j) {
    int tp = t - j;
    if (tp >= 0) {
      float4 u = *(const float4*)(x1 + ((size_t)tp * Nn + n) * CC + c);
      v.x += dk[(c + 0) * KKD + j] * u.x;
      v.y += dk[(c + 1) * KKD + j] * u.y;
      v.z += dk[(c + 2) * KKD + j] * u.z;
      v.w += dk[(c + 3) * KKD + j] * u.w;
    }
  }
  float s  = waveRed(v.x + v.y + v.z + v.w);
  float sq = waveRed(v.x * v.x + v.y * v.y + v.z * v.z + v.w * v.w);
  float mean = s * (1.0f / CC);
  float rstd = rsqrtf(sq * (1.0f / CC) - mean * mean + 1e-5f);
  float4 gg = ((const float4*)g)[lane];
  float4 bb = ((const float4*)be)[lane];
  float4 o;
  o.x = (v.x - mean) * rstd * gg.x + bb.x;
  o.y = (v.y - mean) * rstd * gg.y + bb.y;
  o.z = (v.z - mean) * rstd * gg.z + bb.z;
  o.w = (v.w - mean) * rstd * gg.w + bb.w;
  ((float4*)(out + (size_t)wid * CC))[lane] = o;
}

// ------------- time-windowed attention, thread per (t,n,h); out may alias q ---
__global__ void kattn(const float* __restrict__ q, const float* __restrict__ k,
                      const float* __restrict__ v, const int* __restrict__ tidx,
                      float* __restrict__ out, int Nn) {
  int idx = blockIdx.x * blockDim.x + threadIdx.x;
  if (idx >= TT * Nn * HH) return;
  int h = idx & (HH - 1);
  int n = (idx / HH) % Nn;
  int t = idx / (HH * Nn);
  size_t base = ((size_t)t * Nn + n) * CC + h * DHD;
  const float rs = 0.17677669529663687f;  // 1/sqrt(32)
  float qv[DHD];
#pragma unroll
  for (int d = 0; d < DHD; ++d) qv[d] = q[base + d];
  float sc[WWIN];
#pragma unroll
  for (int w = 0; w < WWIN; ++w) {
    int tp = t - w;
    float s = -1e9f;
    if (tp >= 0 && (tidx[t] - tidx[tp]) < WWIN) {
      size_t kb = ((size_t)tp * Nn + n) * CC + h * DHD;
      float acc = 0.f;
#pragma unroll
      for (int d = 0; d < DHD; ++d) acc += qv[d] * k[kb + d];
      s = acc * rs;
    }
    sc[w] = s;
  }
  float m = fmaxf(fmaxf(sc[0], sc[1]), fmaxf(sc[2], sc[3]));
  float ex[WWIN], se = 0.f;
#pragma unroll
  for (int w = 0; w < WWIN; ++w) { ex[w] = expf(sc[w] - m); se += ex[w]; }
  float inv = 1.0f / se;
#pragma unroll
  for (int w = 0; w < WWIN; ++w) ex[w] *= inv;
  float av[DHD];
#pragma unroll
  for (int d = 0; d < DHD; ++d) av[d] = 0.f;
#pragma unroll
  for (int w = 0; w < WWIN; ++w) {
    int tp = t - w;
    if (tp >= 0) {
      size_t vb = ((size_t)tp * Nn + n) * CC + h * DHD;
      float ww = ex[w];
#pragma unroll
      for (int d = 0; d < DHD; ++d) av[d] += ww * v[vb + d];
    }
  }
#pragma unroll
  for (int d = 0; d < DHD; ++d) out[base + d] = av[d];
}

// ------------- LIF scan over T, wave per node; ffn_in may alias agg ----------
__global__ void klif(const float* __restrict__ agg, float* __restrict__ ffn_in,
                     float* __restrict__ spkMean, int Nn) {
  int wid = blockIdx.x * (blockDim.x >> 5) + (threadIdx.x >> 5);
  if (wid >= Nn) return;
  int lane = threadIdx.x & 31;
  float4 vm = make_float4(0.f, 0.f, 0.f, 0.f);
#pragma unroll
  for (int t = 0; t < TT; ++t) {
    size_t off = ((size_t)t * Nn + wid) * CC + lane * 4;
    float4 in = *(const float4*)(agg + off);
    float4 vn, s, fi;
    vn.x = TAUF * vm.x + in.x; vn.y = TAUF * vm.y + in.y;
    vn.z = TAUF * vm.z + in.z; vn.w = TAUF * vm.w + in.w;
    s.x = (vn.x >= VTHF) ? 1.f : 0.f; s.y = (vn.y >= VTHF) ? 1.f : 0.f;
    s.z = (vn.z >= VTHF) ? 1.f : 0.f; s.w = (vn.w >= VTHF) ? 1.f : 0.f;
    vm.x = vn.x * (1.f - s.x); vm.y = vn.y * (1.f - s.y);
    vm.z = vn.z * (1.f - s.z); vm.w = vn.w * (1.f - s.w);
    fi.x = in.x * s.x; fi.y = in.y * s.y; fi.z = in.z * s.z; fi.w = in.w * s.w;
    *(float4*)(ffn_in + off) = fi;
    float tot = waveRed(s.x + s.y + s.z + s.w);
    if (lane == 0) spkMean[t * Nn + wid] = tot * (1.0f / CC);
  }
}

// ------------- fused FFN: gelu(A@W1+b1)@W2+b2, hidden tile stays in LDS ------
__global__ void kffn(const float* __restrict__ A, const unsigned short* __restrict__ W1t,
                     const float* __restrict__ b1, const unsigned short* __restrict__ W2t,
                     const float* __restrict__ b2, float* __restrict__ out, int M) {
  __shared__ unsigned short aT[16 * 136];
  __shared__ unsigned short hT[16 * 536];
  int rowBase = blockIdx.x * 16;
  for (int i = threadIdx.x; i < 16 * CC; i += blockDim.x) {
    int r = i >> 7, c = i & 127;
    aT[r * 136 + c] = f2bf_bits(A[(size_t)(rowBase + r) * CC + c]);
  }
  __syncthreads();
  int wave = threadIdx.x >> 5, lane = threadIdx.x & 31;
  int nlo = lane & 15, rOff = (lane & 16) ? 8 : 0;
  // stage 1: hid = gelu(A @ W1 + b1), 512 cols (64 per wave)
#pragma unroll
  for (int ct = 0; ct < 4; ++ct) {
    int c0 = wave * 64 + ct * 16;
    v8f acc = {};
#pragma unroll
    for (int k0 = 0; k0 < CC; k0 += 32) {
      v16bf a = load_frag(aT, 136, k0, lane);
      v16bf b = load_frag(W1t + (size_t)c0 * CC, CC, k0, lane);
      acc = __builtin_amdgcn_wmma_f32_16x16x32_bf16(false, a, false, b, (short)0, acc,
                                                    false, false);
    }
    int col = c0 + nlo;
    float bb = b1[col];
#pragma unroll
    for (int r = 0; r < 8; ++r)
      hT[(rOff + r) * 536 + col] = f2bf_bits(gelu_tanh(acc[r] + bb));
  }
  __syncthreads();
  // stage 2: out = hid @ W2 + b2, K=512
  int c0 = wave * 16;
  v8f acc = {};
#pragma unroll
  for (int k0 = 0; k0 < 512; k0 += 32) {
    v16bf a = load_frag(hT, 536, k0, lane);
    v16bf b = load_frag(W2t + (size_t)c0 * 512, 512, k0, lane);
    acc = __builtin_amdgcn_wmma_f32_16x16x32_bf16(false, a, false, b, (short)0, acc,
                                                  false, false);
  }
  int col = c0 + nlo;
  float bb = b2[col];
#pragma unroll
  for (int r = 0; r < 8; ++r)
    out[(size_t)(rowBase + rOff + r) * CC + col] = acc[r] + bb;
}

// ---------------- host orchestration ----------------

extern "C" void kernel_launch(void* const* d_in, const int* in_sizes, int n_in,
                              void* d_out, int out_size, void* d_ws, size_t ws_size,
                              hipStream_t stream) {
  const float* x      = (const float*)d_in[0];
  const float* spikes = (const float*)d_in[1];
  const int*   eidx   = (const int*)d_in[2];
  const int*   tidx   = (const int*)d_in[3];
  const float* Ws  = (const float*)d_in[4];
  const float* bs  = (const float*)d_in[5];
  const float* dk  = (const float*)d_in[6];
  const float* g1  = (const float*)d_in[7];
  const float* b1  = (const float*)d_in[8];
  const float* g2  = (const float*)d_in[9];
  const float* b2  = (const float*)d_in[10];
  const float* gf  = (const float*)d_in[11];
  const float* bf_ = (const float*)d_in[12];
  const float* Wq = (const float*)d_in[13]; const float* bq = (const float*)d_in[14];
  const float* Wk = (const float*)d_in[15]; const float* bk = (const float*)d_in[16];
  const float* Wv = (const float*)d_in[17]; const float* bv = (const float*)d_in[18];
  const float* Wo = (const float*)d_in[19]; const float* bo = (const float*)d_in[20];
  const float* fc1w = (const float*)d_in[21]; const float* fc1b = (const float*)d_in[22];
  const float* fc2w = (const float*)d_in[23]; const float* fc2b = (const float*)d_in[24];

  const int Nn = in_sizes[1] / TT;   // 20000
  const int E  = in_sizes[2] / 2;    // 320000
  const int M  = TT * Nn;            // 160000 rows
  const size_t SZ = (size_t)M * CC;  // floats per activation slot

  const int* srcp = eidx;
  const int* dstp = eidx + E;

  float* fp = (float*)d_ws;
  float* inv_deg = fp;
  size_t o = ((size_t)Nn + 255) & ~(size_t)255;
  float* S0 = fp + o;        // scatter accumulator
  float* SA = S0 + SZ;       // x_spatial / q / attn-out / aggregated / ffn_in / ffn_out
  float* SB = SA + SZ;       // x1 / k
  float* SC = SB + SZ;       // x2
  float* SD = SC + SZ;       // v
  unsigned short* wb  = (unsigned short*)(SD + SZ);
  unsigned short* WsT = wb;
  unsigned short* WqT = WsT + 16384;
  unsigned short* WkT = WqT + 16384;
  unsigned short* WvT = WkT + 16384;
  unsigned short* WoT = WvT + 16384;
  unsigned short* W1T = WoT + 16384;  // [512,128]
  unsigned short* W2T = W1T + 65536;  // [128,512]

  const int B = 256;
  const int rowTiles = M / 16;                 // 10000
  const int waveBlksRows = (M + 7) / 8;        // 8 waves/block
  const long long scWaves = (long long)TT * E; // 2.56M
  const int scBlks = (int)((scWaves + 7) / 8);

  // weights -> transposed bf16
  kw2bf<<<(16384 + B - 1) / B, B, 0, stream>>>(Ws, WsT, CC, CC);
  kw2bf<<<(16384 + B - 1) / B, B, 0, stream>>>(Wq, WqT, CC, CC);
  kw2bf<<<(16384 + B - 1) / B, B, 0, stream>>>(Wk, WkT, CC, CC);
  kw2bf<<<(16384 + B - 1) / B, B, 0, stream>>>(Wv, WvT, CC, CC);
  kw2bf<<<(16384 + B - 1) / B, B, 0, stream>>>(Wo, WoT, CC, CC);
  kw2bf<<<(65536 + B - 1) / B, B, 0, stream>>>(fc1w, W1T, CC, 4 * CC);
  kw2bf<<<(65536 + B - 1) / B, B, 0, stream>>>(fc2w, W2T, 4 * CC, CC);

  // degrees -> inv_deg
  kzero<<<(Nn / 4 + B - 1) / B, B, 0, stream>>>(inv_deg, Nn / 4);
  kdeg<<<(E + B - 1) / B, B, 0, stream>>>(dstp, inv_deg, E);
  kinv<<<(Nn + B - 1) / B, B, 0, stream>>>(inv_deg, Nn);

  // spatial GNN: scatter-mean(x) @ Ws + bs ; x1 = LN(x + xsp)
  kzero<<<(int)(SZ / 4 + B - 1) / B, B, 0, stream>>>(S0, SZ / 4);
  kscatter<<<scBlks, B, 0, stream>>>(x, srcp, dstp, nullptr, S0, E, Nn);
  kgemm128<<<rowTiles, B, 0, stream>>>(S0, WsT, bs, inv_deg, Nn, SA, M);
  klnres<<<waveBlksRows, B, 0, stream>>>(x, SA, g1, b1, SB, M);

  // temporal delay line + LN -> x2
  kdelayln<<<waveBlksRows, B, 0, stream>>>(SB, dk, g2, b2, SC, Nn);

  // gated scatter-mean -> m; q/k/v projections
  kzero<<<(int)(SZ / 4 + B - 1) / B, B, 0, stream>>>(S0, SZ / 4);
  kscatter<<<scBlks, B, 0, stream>>>(SC, srcp, dstp, spikes, S0, E, Nn);
  kgemm128<<<rowTiles, B, 0, stream>>>(SC, WqT, bq, nullptr, 1, SA, M);      // q
  kgemm128<<<rowTiles, B, 0, stream>>>(S0, WkT, bk, inv_deg, Nn, SB, M);     // k
  kgemm128<<<rowTiles, B, 0, stream>>>(S0, WvT, bv, inv_deg, Nn, SD, M);     // v

  // windowed attention (in-place into q slot), then Wo projection (in-place)
  kattn<<<(TT * Nn * HH + B - 1) / B, B, 0, stream>>>(SA, SB, SD, tidx, SA, Nn);
  kgemm128<<<rowTiles, B, 0, stream>>>(SA, WoT, bo, nullptr, 1, SA, M);      // aggregated

  // LIF scan: spikes mean -> d_out tail; ffn_in = aggregated*spk (in-place)
  klif<<<(Nn + 7) / 8, B, 0, stream>>>(SA, SA, (float*)d_out + SZ, Nn);

  // fused FFN (in-place), then final residual LN -> d_out head
  kffn<<<rowTiles, B, 0, stream>>>(SA, W1T, fc1b, W2T, fc2b, SA, M);
  klnres<<<waveBlksRows, B, 0, stream>>>(SC, SA, gf, bf_, (float*)d_out, M);
}
